// PaiNNModel_79663053406274
// MI455X (gfx1250) — matrix operand order repaired and checked by reading, not compiled, not run on hardware
//
#include <hip/hip_runtime.h>
#include <hip/hip_bf16.h>

#define NN   50000
#define EE   800000
#define FF   64
#define F3   192
#define RR   20
#define LL   2
#define BB   64
#define RCUT 5.0f
#define LN2F 0.69314718056f
#define PIF  3.14159265358979f

typedef __attribute__((ext_vector_type(16))) _Float16 v16h;
typedef __attribute__((ext_vector_type(8)))  _Float16 v8h;
typedef __attribute__((ext_vector_type(8)))  float    v8f;

static __device__ __forceinline__ float sspf(float x) {
    // softplus(x) - ln2, numerically stable
    float ax = fabsf(x);
    return fmaxf(x, 0.0f) + log1pf(expf(-ax)) - LN2F;
}

static __device__ __forceinline__ v8f splat8(float x) {
    v8f r;
#pragma unroll
    for (int i = 0; i < 8; i++) r[i] = x;
    return r;
}

static __device__ __forceinline__ v8f wmma_f16(v16h a, v16h b, v8f c) {
    return __builtin_amdgcn_wmma_f32_16x16x32_f16(false, a, false, b, (short)0, c, false, false);
}

// A fragment: 16xK row-major f16 tile in LDS (row stride in halves).
// Lane layout (16b A 16x32): lanes 0-15 -> K=0..7,16..23 ; lanes 16-31 -> K=8..15,24..31.
static __device__ __forceinline__ v16h load_a_frag(const _Float16* tile, int stride, int lane, int kstep) {
    int row = lane & 15;
    int kb  = kstep * 32 + ((lane & 16) ? 8 : 0);
    const _Float16* p = tile + row * stride + kb;
    v8h lo = *(const v8h*)p;
    v8h hi = *(const v8h*)(p + 16);
    v16h a;
#pragma unroll
    for (int i = 0; i < 8; i++) { a[i] = lo[i]; a[i + 8] = hi[i]; }
    return a;
}

// B fragment from pre-packed weights: pack[((ntile*ksteps)+kstep)*32 + lane][0..15]
static __device__ __forceinline__ v16h load_b_frag(const _Float16* pack, int ntile, int kstep, int ksteps, int lane) {
    return *(const v16h*)(pack + (size_t)(((ntile * ksteps) + kstep) * 32 + lane) * 16);
}

// ---------------- weight packing: W (Out x K) row-major f32 -> B-frag f16 layout ----------------
__global__ void pack_b_kernel(const float* __restrict__ src, _Float16* __restrict__ dst, int Out, int K) {
    int t    = blockIdx.x * blockDim.x + threadIdx.x;
    int lane = t & 31;
    int frag = t >> 5;
    int ksteps = K / 32;
    int nfrags = (Out / 16) * ksteps;
    if (frag >= nfrags) return;
    int ntile = frag / ksteps;
    int kstep = frag % ksteps;
    int n  = ntile * 16 + (lane & 15);
    int kb = kstep * 32 + ((lane & 16) ? 8 : 0);
    _Float16* d = dst + (size_t)(frag * 32 + lane) * 16;
#pragma unroll
    for (int i = 0; i < 16; i++) d[i] = (_Float16)src[n * K + kb + i + ((i & 8) ? 8 : 0)];
}

// ---------------- edge geometry ----------------
__global__ void prep_edges_kernel(const float* __restrict__ pos, const int* __restrict__ ei,
                                  float* __restrict__ rhat, float* __restrict__ rnorm) {
    int e = blockIdx.x * blockDim.x + threadIdx.x;
    if (e >= EE) return;
    int src = ei[e];
    int dst = ei[EE + e];
    float rx = pos[dst * 3 + 0] - pos[src * 3 + 0];
    float ry = pos[dst * 3 + 1] - pos[src * 3 + 1];
    float rz = pos[dst * 3 + 2] - pos[src * 3 + 2];
    float rn = sqrtf(rx * rx + ry * ry + rz * rz);
    float inv = 1.0f / fmaxf(rn, 1e-8f);
    rhat[e * 3 + 0] = rx * inv;
    rhat[e * 3 + 1] = ry * inv;
    rhat[e * 3 + 2] = rz * inv;
    rnorm[e] = rn;
}

// ---------------- state init ----------------
__global__ void init_state_kernel(const int* __restrict__ z, const float* __restrict__ emb,
                                  float* __restrict__ s, float* __restrict__ v) {
    int idx = blockIdx.x * blockDim.x + threadIdx.x;  // n*FF + f
    if (idx >= NN * FF) return;
    int n = idx >> 6, f = idx & 63;
    s[idx] = emb[z[n] * FF + f];
    v[idx * 3 + 0] = 0.0f;
    v[idx * 3 + 1] = 0.0f;
    v[idx * 3 + 2] = 0.0f;
}

// ---------------- per-layer edge message block (WMMA) ----------------
// wave handles 16 edges; block = 4 waves = 64 edges; grid = EE/64 = 12500
__global__ __launch_bounds__(128) void edge_message_kernel(
    const float* __restrict__ s_in, const float* __restrict__ v_in,
    float* __restrict__ s_acc, float* __restrict__ v_acc,
    const int* __restrict__ ei,
    const float* __restrict__ rnorm, const float* __restrict__ rhat,
    const float* __restrict__ fW1_l, const float* __restrict__ fb1_l,
    const float* __restrict__ fb2_l, const float* __restrict__ phib_l,
    const _Float16* __restrict__ phiP_l, const _Float16* __restrict__ fW2P_l) {
    __shared__ __align__(16) _Float16 Ts[4][16 * 64];  // s[dst] tile (f16)
    __shared__ __align__(16) _Float16 Tt[4][16 * 64];  // ssp(rbf @ W1 + b1) tile (f16)
    __shared__ int Eidx[4][2][16];

    int tid = threadIdx.x, w = tid >> 5, lane = tid & 31;
    int ebase = (blockIdx.x * 4 + w) * 16;
    int edge = lane & 15, half = lane >> 4;

    if (lane < 16) {
        Eidx[w][0][lane] = ei[ebase + lane];       // src
        Eidx[w][1][lane] = ei[EE + ebase + lane];  // dst
    }
    __syncthreads();

    // ---- stage A tiles ----
    int dn = Eidx[w][1][edge];
    const float* srow = s_in + dn * FF + half * 32;
#pragma unroll 8
    for (int c = 0; c < 32; c++) Ts[w][edge * 64 + half * 32 + c] = (_Float16)srow[c];

    float rn = rnorm[ebase + edge];
    float fc = (rn < RCUT) ? (0.5f * (cosf(PIF * rn / RCUT) + 1.0f)) : 0.0f;
    const float dd = RCUT / (float)(RR - 1);
    float ev[RR];
#pragma unroll
    for (int r = 0; r < RR; r++) {
        float x = rn - (float)r * dd;
        ev[r] = expf(-x * x / (2.0f * dd * dd)) * fc;
    }
    const float* W1r = fW1_l + (half * 32) * RR;
#pragma unroll 2
    for (int c = 0; c < 32; c++) {
        float acc = fb1_l[half * 32 + c];
#pragma unroll
        for (int r = 0; r < RR; r++) acc += ev[r] * W1r[c * RR + r];
        Tt[w][edge * 64 + half * 32 + c] = (_Float16)sspf(acc);
    }
    __syncthreads();

    // ---- WMMA over 12 output tiles of 16 cols (192 total) ----
    v16h aS0 = load_a_frag(&Ts[w][0], 64, lane, 0);
    v16h aS1 = load_a_frag(&Ts[w][0], 64, lane, 1);
    v16h aT0 = load_a_frag(&Tt[w][0], 64, lane, 0);
    v16h aT1 = load_a_frag(&Tt[w][0], 64, lane, 1);
    int col = lane & 15, rowoff = (lane >> 4) * 8;

    for (int nt = 0; nt < 12; ++nt) {
        v8f cf = splat8(fb2_l[nt * 16 + col]);
        cf = wmma_f16(aT0, load_b_frag(fW2P_l, nt, 0, 2, lane), cf);
        cf = wmma_f16(aT1, load_b_frag(fW2P_l, nt, 1, 2, lane), cf);

        v8f cp = splat8(0.0f);
        if (nt < 4 || nt >= 8) {  // phi needed only for scalar & r-gate thirds
            cp = splat8(phib_l[nt * 16 + col]);
            cp = wmma_f16(aS0, load_b_frag(phiP_l, nt, 0, 2, lane), cp);
            cp = wmma_f16(aS1, load_b_frag(phiP_l, nt, 1, 2, lane), cp);
        }

#pragma unroll
        for (int j = 0; j < 8; j++) {
            int m  = rowoff + j;
            int sn = Eidx[w][0][m];
            if (nt < 4) {
                int g = nt * 16 + col;
                atomicAdd(&s_acc[sn * FF + g], cf[j] * cp[j]);
            } else if (nt < 8) {
                int f = (nt - 4) * 16 + col;
                int dnm = Eidx[w][1][m];
                const float* vd = v_in + (dnm * FF + f) * 3;
                float* vo = v_acc + (sn * FF + f) * 3;
                float wv = cf[j];
                atomicAdd(vo + 0, wv * vd[0]);
                atomicAdd(vo + 1, wv * vd[1]);
                atomicAdd(vo + 2, wv * vd[2]);
            } else {
                int f = (nt - 8) * 16 + col;
                const float* rh = rhat + (ebase + m) * 3;
                float* vo = v_acc + (sn * FF + f) * 3;
                float wr = cf[j] * cp[j];
                atomicAdd(vo + 0, wr * rh[0]);
                atomicAdd(vo + 1, wr * rh[1]);
                atomicAdd(vo + 2, wr * rh[2]);
            }
        }
    }
}

// ---------------- per-layer Uv / Vv GEMMs + norms (WMMA) ----------------
// wave handles 16 nodes; 3125 wave tiles -> 782 blocks of 4 waves
__global__ __launch_bounds__(128) void uv_gemm_kernel(
    const float* __restrict__ v_cur, float* __restrict__ Uv,
    float* __restrict__ normsq, float* __restrict__ inner,
    const _Float16* __restrict__ UP_l, const _Float16* __restrict__ VP_l) {
    __shared__ __align__(16) _Float16 Av[4][3][16 * 64];
    int tid = threadIdx.x, w = tid >> 5, lane = tid & 31;
    int nbase = (blockIdx.x * 4 + w) * 16;
    bool wvalid = (nbase < NN);
    if (!wvalid) nbase = 0;
    int node = lane & 15, half = lane >> 4;

#pragma unroll
    for (int d = 0; d < 3; d++) {
#pragma unroll 8
        for (int c = 0; c < 32; c++) {
            int f = half * 32 + c;
            Av[w][d][node * 64 + f] = (_Float16)v_cur[((nbase + node) * FF + f) * 3 + d];
        }
    }
    __syncthreads();

    int col = lane & 15, rowoff = (lane >> 4) * 8;
    for (int gt = 0; gt < 4; gt++) {
        float nrm[8], inr[8];
#pragma unroll
        for (int j = 0; j < 8; j++) { nrm[j] = 0.0f; inr[j] = 0.0f; }
#pragma unroll
        for (int d = 0; d < 3; d++) {
            v8f cu = splat8(0.0f), cv = splat8(0.0f);
#pragma unroll
            for (int ks = 0; ks < 2; ks++) {
                v16h a = load_a_frag(&Av[w][d][0], 64, lane, ks);
                cu = wmma_f16(a, load_b_frag(UP_l, gt, ks, 2, lane), cu);
                cv = wmma_f16(a, load_b_frag(VP_l, gt, ks, 2, lane), cv);
            }
#pragma unroll
            for (int j = 0; j < 8; j++) {
                int n = nbase + rowoff + j;
                int g = gt * 16 + col;
                if (wvalid) Uv[(n * FF + g) * 3 + d] = cu[j];
                nrm[j] += cv[j] * cv[j];
                inr[j] += cu[j] * cv[j];
            }
        }
#pragma unroll
        for (int j = 0; j < 8; j++) {
            if (wvalid) {
                int idx = (nbase + rowoff + j) * FF + gt * 16 + col;
                normsq[idx] = nrm[j];
                inner[idx]  = inr[j];
            }
        }
    }
}

// ---------------- per-layer node a-MLP + state update (WMMA) ----------------
__global__ __launch_bounds__(128) void node_mlp_kernel(
    float* __restrict__ s_cur, float* __restrict__ v_cur,
    const float* __restrict__ Uv, const float* __restrict__ normsq, const float* __restrict__ inner,
    const _Float16* __restrict__ aW1P_l, const _Float16* __restrict__ aW2P_l,
    const float* __restrict__ ab1_l, const float* __restrict__ ab2_l) {
    __shared__ __align__(16) _Float16 A1[4][16 * 128];  // [s | ||Vv||^2]
    __shared__ __align__(16) _Float16 H[4][16 * 64];    // hidden
    int tid = threadIdx.x, w = tid >> 5, lane = tid & 31;
    int nbase = (blockIdx.x * 4 + w) * 16;
    bool wvalid = (nbase < NN);
    if (!wvalid) nbase = 0;
    int node = lane & 15, half = lane >> 4;

#pragma unroll 8
    for (int c = 0; c < 64; c++) {
        int cc = half * 64 + c;
        float val = (cc < 64) ? s_cur[(nbase + node) * FF + cc]
                              : normsq[(nbase + node) * FF + (cc - 64)];
        A1[w][node * 128 + cc] = (_Float16)val;
    }
    __syncthreads();

    int col = lane & 15, rowoff = (lane >> 4) * 8;
    for (int gt = 0; gt < 4; gt++) {
        v8f ch = splat8(ab1_l[gt * 16 + col]);
#pragma unroll
        for (int ks = 0; ks < 4; ks++)
            ch = wmma_f16(load_a_frag(&A1[w][0], 128, lane, ks),
                          load_b_frag(aW1P_l, gt, ks, 4, lane), ch);
#pragma unroll
        for (int j = 0; j < 8; j++)
            H[w][(rowoff + j) * 64 + gt * 16 + col] = (_Float16)sspf(ch[j]);
    }
    __syncthreads();

    for (int gt = 0; gt < 4; gt++) {
        v8f css = splat8(ab2_l[gt * 16 + col]);
        v8f csv = splat8(ab2_l[64 + gt * 16 + col]);
        v8f cvv = splat8(ab2_l[128 + gt * 16 + col]);
#pragma unroll
        for (int ks = 0; ks < 2; ks++) {
            v16h a = load_a_frag(&H[w][0], 64, lane, ks);
            css = wmma_f16(a, load_b_frag(aW2P_l, gt, ks, 2, lane), css);
            csv = wmma_f16(a, load_b_frag(aW2P_l, gt + 4, ks, 2, lane), csv);
            cvv = wmma_f16(a, load_b_frag(aW2P_l, gt + 8, ks, 2, lane), cvv);
        }
#pragma unroll
        for (int j = 0; j < 8; j++) {
            if (!wvalid) continue;
            int idx = (nbase + rowoff + j) * FF + gt * 16 + col;
            s_cur[idx] += css[j] + csv[j] * inner[idx];
            const float* uv = Uv + idx * 3;
            v_cur[idx * 3 + 0] += cvv[j] * uv[0];
            v_cur[idx * 3 + 1] += cvv[j] * uv[1];
            v_cur[idx * 3 + 2] += cvv[j] * uv[2];
        }
    }
}

// ---------------- heads ----------------
__global__ void heads_kernel(const float* __restrict__ s, const float* __restrict__ v,
                             const int* __restrict__ batch,
                             const float* __restrict__ eW1, const float* __restrict__ eb1,
                             const float* __restrict__ eW2, const float* __restrict__ eb2,
                             const float* __restrict__ dipw,
                             float* __restrict__ eacc, float* __restrict__ muacc,
                             float* __restrict__ charges) {
    int n = blockIdx.x * blockDim.x + threadIdx.x;
    if (n >= NN) return;
    const float* sr = s + n * FF;
    float eps = eb2[0];
#pragma unroll 2
    for (int h = 0; h < FF / 2; h++) {
        float a = eb1[h];
#pragma unroll 8
        for (int f = 0; f < FF; f++) a += sr[f] * eW1[h * FF + f];
        eps += eW2[h] * sspf(a);
    }
    int b = batch[n];
    atomicAdd(&eacc[b], eps);
#pragma unroll
    for (int d = 0; d < 3; d++) {
        float m = 0.0f;
#pragma unroll 8
        for (int f = 0; f < FF; f++) m += v[(n * FF + f) * 3 + d] * dipw[f];
        atomicAdd(&muacc[b * 3 + d], m);
    }
    charges[n] = 0.0f;
}

__global__ void finalize_kernel(const float* __restrict__ eacc, const float* __restrict__ muacc,
                                float* __restrict__ out) {
    int b = threadIdx.x;
    if (b < BB) {
        out[b] = eacc[b];
        float x = muacc[b * 3 + 0], y = muacc[b * 3 + 1], z = muacc[b * 3 + 2];
        out[BB + b] = sqrtf(x * x + y * y + z * z);
    }
}

// ---------------- host launcher ----------------
extern "C" void kernel_launch(void* const* d_in, const int* in_sizes, int n_in,
                              void* d_out, int out_size, void* d_ws, size_t ws_size,
                              hipStream_t stream) {
    const int*   z     = (const int*)  d_in[0];
    const float* pos   = (const float*)d_in[1];
    const int*   ei    = (const int*)  d_in[2];
    const int*   batch = (const int*)  d_in[3];
    const float* emb   = (const float*)d_in[4];
    const float* phi_W = (const float*)d_in[5];
    const float* phi_b = (const float*)d_in[6];
    const float* fW1   = (const float*)d_in[7];
    const float* fb1   = (const float*)d_in[8];
    const float* fW2   = (const float*)d_in[9];
    const float* fb2   = (const float*)d_in[10];
    const float* U_W   = (const float*)d_in[11];
    const float* V_W   = (const float*)d_in[12];
    const float* aW1   = (const float*)d_in[13];
    const float* ab1   = (const float*)d_in[14];
    const float* aW2   = (const float*)d_in[15];
    const float* ab2   = (const float*)d_in[16];
    const float* eW1   = (const float*)d_in[17];
    const float* eb1   = (const float*)d_in[18];
    const float* eW2   = (const float*)d_in[19];
    const float* eb2   = (const float*)d_in[20];
    const float* dipw  = (const float*)d_in[21];
    float* out = (float*)d_out;

    char* base = (char*)d_ws;
    size_t off = 0;
    auto take = [&](size_t bytes) -> char* {
        char* p = base + off;
        off = (off + bytes + 255) & ~(size_t)255;
        return p;
    };
    float* s_a    = (float*)take((size_t)NN * FF * 4);
    float* s_b    = (float*)take((size_t)NN * FF * 4);
    float* v_a    = (float*)take((size_t)NN * FF * 3 * 4);
    float* v_b    = (float*)take((size_t)NN * FF * 3 * 4);
    float* rhat   = (float*)take((size_t)EE * 3 * 4);
    float* rnorm  = (float*)take((size_t)EE * 4);
    float* Uv     = (float*)take((size_t)NN * FF * 3 * 4);
    float* normsq = (float*)take((size_t)NN * FF * 4);
    float* inner  = (float*)take((size_t)NN * FF * 4);
    float* eacc   = (float*)take((size_t)(BB + 3 * BB) * 4);
    float* muacc  = eacc + BB;

    const size_t phiP_sz = 12 * 2 * 32 * 16;  // halves
    const size_t fW2P_sz = 12 * 2 * 32 * 16;
    const size_t UP_sz   = 4 * 2 * 32 * 16;
    const size_t VP_sz   = 4 * 2 * 32 * 16;
    const size_t aW1P_sz = 4 * 4 * 32 * 16;
    const size_t aW2P_sz = 12 * 2 * 32 * 16;
    _Float16* phiP  = (_Float16*)take(LL * phiP_sz * 2);
    _Float16* fW2P  = (_Float16*)take(LL * fW2P_sz * 2);
    _Float16* UP    = (_Float16*)take(LL * UP_sz * 2);
    _Float16* VP    = (_Float16*)take(LL * VP_sz * 2);
    _Float16* aW1P  = (_Float16*)take(LL * aW1P_sz * 2);
    _Float16* aW2P  = (_Float16*)take(LL * aW2P_sz * 2);

    // ---- pack weights into WMMA B-fragment layout ----
    for (int l = 0; l < LL; l++) {
        pack_b_kernel<<<(12 * 2 * 32 + 255) / 256, 256, 0, stream>>>(phi_W + l * F3 * FF, phiP + l * phiP_sz, F3, FF);
        pack_b_kernel<<<(12 * 2 * 32 + 255) / 256, 256, 0, stream>>>(fW2 + l * F3 * FF, fW2P + l * fW2P_sz, F3, FF);
        pack_b_kernel<<<(4 * 2 * 32 + 255) / 256, 256, 0, stream>>>(U_W + l * FF * FF, UP + l * UP_sz, FF, FF);
        pack_b_kernel<<<(4 * 2 * 32 + 255) / 256, 256, 0, stream>>>(V_W + l * FF * FF, VP + l * VP_sz, FF, FF);
        pack_b_kernel<<<(4 * 4 * 32 + 255) / 256, 256, 0, stream>>>(aW1 + l * FF * 2 * FF, aW1P + l * aW1P_sz, FF, 2 * FF);
        pack_b_kernel<<<(12 * 2 * 32 + 255) / 256, 256, 0, stream>>>(aW2 + l * F3 * FF, aW2P + l * aW2P_sz, F3, FF);
    }

    prep_edges_kernel<<<(EE + 255) / 256, 256, 0, stream>>>(pos, ei, rhat, rnorm);
    init_state_kernel<<<(NN * FF + 255) / 256, 256, 0, stream>>>(z, emb, s_a, v_a);

    float* scur = s_a; float* snew = s_b;
    float* vcur = v_a; float* vnew = v_b;
    const int node_blocks = (NN / 16 + 3) / 4;  // 3125 wave tiles -> 782 blocks

    for (int l = 0; l < LL; l++) {
        hipMemcpyAsync(snew, scur, (size_t)NN * FF * 4, hipMemcpyDeviceToDevice, stream);
        hipMemcpyAsync(vnew, vcur, (size_t)NN * FF * 3 * 4, hipMemcpyDeviceToDevice, stream);

        edge_message_kernel<<<EE / 64, 128, 0, stream>>>(
            scur, vcur, snew, vnew, ei, rnorm, rhat,
            fW1 + l * FF * RR, fb1 + l * FF, fb2 + l * F3, phi_b + l * F3,
            phiP + l * phiP_sz, fW2P + l * fW2P_sz);

        uv_gemm_kernel<<<node_blocks, 128, 0, stream>>>(
            vnew, Uv, normsq, inner, UP + l * UP_sz, VP + l * VP_sz);

        node_mlp_kernel<<<node_blocks, 128, 0, stream>>>(
            snew, vnew, Uv, normsq, inner,
            aW1P + l * aW1P_sz, aW2P + l * aW2P_sz, ab1 + l * FF, ab2 + l * F3);

        float* ts = scur; scur = snew; snew = ts;
        float* tv = vcur; vcur = vnew; vnew = tv;
    }

    hipMemsetAsync(eacc, 0, (size_t)(BB + 3 * BB) * 4, stream);
    heads_kernel<<<(NN + 255) / 256, 256, 0, stream>>>(
        scur, vcur, batch, eW1, eb1, eW2, eb2, dipw, eacc, muacc, out + 2 * BB);
    finalize_kernel<<<1, 64, 0, stream>>>(eacc, muacc, out);
}
